// DynamicFeedForward_36266703847578
// MI455X (gfx1250) — matrix operand order repaired and checked by reading, not compiled, not run on hardware
//
#include <hip/hip_runtime.h>
#include <hip/hip_bf16.h>

typedef __attribute__((ext_vector_type(16))) __bf16        v16bf;
typedef __attribute__((ext_vector_type(8)))  float         v8f;
typedef __attribute__((ext_vector_type(4)))  unsigned int  v4u;

#define B_    8
#define CIN   64
#define DIMS_ 64
#define NW_   8
#define T_    16
#define H_    56
#define W_    56
#define HID_  17
#define KTOT  1728            // CIN * 27
#define PC    50176           // T*H*W per (b,c)
#define NELEM 25690112        // B*DIMS*PC
#define NRED  401408.0f       // B*PC

// workspace layout (bytes), all 16B-aligned
static constexpr size_t OFF_Y    = 0;
static constexpr size_t OFF_DYNW = (size_t)NELEM * 4;          // y: 102,760,448
static constexpr size_t OFF_POOL = OFF_DYNW + 884736ull * 2;   // dyn_w bf16
static constexpr size_t OFF_PHI  = OFF_POOL + 2048;
static constexpr size_t OFF_DYNB = OFF_PHI + 256;
static constexpr size_t OFF_STAT = OFF_DYNB + 2048;
static constexpr size_t OFF_XT   = OFF_STAT + 512;             // x bf16, [b][t][h][w][ci]

__device__ __forceinline__ unsigned short f2bf(float f) {
  union { float f; unsigned u; } v; v.f = f;
  return (unsigned short)((v.u + 0x7FFFu + ((v.u >> 16) & 1u)) >> 16);
}

__global__ void k_init(float* stats) {
  if (threadIdx.x < 128) stats[threadIdx.x] = 0.f;
}

// one block per (b,c): max over 50176 contiguous floats
__global__ void k_maxpool(const float* __restrict__ x, float* __restrict__ pooled) {
  __shared__ float red[256];
  const float4* p = (const float4*)(x + (size_t)blockIdx.x * PC);
  float m = -3.4e38f;
  for (int i = threadIdx.x; i < PC / 4; i += 256) {
    float4 v = p[i];
    m = fmaxf(m, fmaxf(fmaxf(v.x, v.y), fmaxf(v.z, v.w)));
  }
  red[threadIdx.x] = m;
  __syncthreads();
  for (int s = 128; s > 0; s >>= 1) {
    if ((int)threadIdx.x < s) red[threadIdx.x] = fmaxf(red[threadIdx.x], red[threadIdx.x + s]);
    __syncthreads();
  }
  if (threadIdx.x == 0) pooled[blockIdx.x] = red[0];
}

// transpose + convert: x (f32, w contiguous) -> xT (bf16, ci contiguous)
__global__ void k_xt(const float* __restrict__ x, unsigned short* __restrict__ xt) {
  __shared__ unsigned short tile[64 * 56];
  const int bid = blockIdx.x;              // b*T*H planes
  const int hy  = bid % H_;
  const int tz  = (bid / H_) % T_;
  const int bb  = bid / (H_ * T_);
  for (int i = threadIdx.x; i < 64 * 56; i += 256) {
    const int ci = i / 56, w = i % 56;     // coalesced in w
    tile[ci * 56 + w] =
        f2bf(x[((((size_t)bb * 64 + ci) * T_ + tz) * H_ + hy) * W_ + w]);
  }
  __syncthreads();
  unsigned short* dst = xt + (((size_t)bb * T_ + tz) * H_ + hy) * (size_t)(W_ * 64);
  for (int i = threadIdx.x; i < 64 * 56; i += 256) {
    const int w = i >> 6, ci = i & 63;     // coalesced in ci
    dst[i] = tile[ci * 56 + w];
  }
}

// tiny MLP + softmax(tau) + dyn_b ; single block of 64 threads
__global__ void k_mlp(const float* __restrict__ pooled,
                      const float* __restrict__ W1, const float* __restrict__ b1,
                      const float* __restrict__ W2, const float* __restrict__ b2,
                      const float* __restrict__ biases, const int* __restrict__ epochs,
                      float* __restrict__ phi, float* __restrict__ dynb) {
  __shared__ float sphi[B_ * NW_];
  const int tid = threadIdx.x;
  if (tid < B_) {
    float h[HID_];
    for (int j = 0; j < HID_; ++j) {
      float s = b1[j];
      for (int c = 0; c < CIN; ++c) s += pooled[tid * CIN + c] * W1[j * CIN + c];
      h[j] = fmaxf(s, 0.f);
    }
    const int ep = epochs[0];
    const float tau = (ep < 10) ? (30.0f - 2.9f * (float)ep) : 1.0f;
    float lg[NW_], mx = -3.4e38f;
    for (int n = 0; n < NW_; ++n) {
      float s = b2[n];
      for (int j = 0; j < HID_; ++j) s += h[j] * W2[n * HID_ + j];
      lg[n] = s / tau;
      mx = fmaxf(mx, lg[n]);
    }
    float den = 0.f;
    for (int n = 0; n < NW_; ++n) { lg[n] = __expf(lg[n] - mx); den += lg[n]; }
    for (int n = 0; n < NW_; ++n) {
      float p = lg[n] / den;
      sphi[tid * NW_ + n] = p;
      phi[tid * NW_ + n]  = p;
    }
  }
  __syncthreads();
  for (int i = tid; i < B_ * DIMS_; i += blockDim.x) {
    const int bb = i / DIMS_, oc = i % DIMS_;
    float s = 0.f;
    for (int n = 0; n < NW_; ++n) s += sphi[bb * NW_ + n] * biases[n * DIMS_ + oc];
    dynb[i] = s;
  }
}

// dyn_w[b][oc][tap][ci] (bf16) = sum_n phi[b,n] * weights[n,oc,ci,dz,dy,dx]
__global__ void k_dynw(const float* __restrict__ weights, const float* __restrict__ phi,
                       unsigned short* __restrict__ dynw) {
  const int idx = blockIdx.x * 256 + threadIdx.x;     // exactly 884736 threads
  const int ci  = idx & 63;
  const int tap = (idx >> 6) % 27;
  const int oc  = (idx / (64 * 27)) & 63;
  const int bb  = idx / (64 * 27 * 64);
  float s = 0.f;
  for (int n = 0; n < NW_; ++n)
    s += phi[bb * NW_ + n] * weights[((size_t)(n * 64 + oc) * 64 + ci) * 27 + tap];
  dynw[idx] = f2bf(s);
}

// implicit-GEMM conv3d: block = 64 oc x 32 w positions at fixed (b,t,h)
__global__ __launch_bounds__(256, 2)
void k_conv(const unsigned short* __restrict__ xt, const unsigned short* __restrict__ dynw,
            const float* __restrict__ dynb, float* __restrict__ y) {
  // slab layout: [tap9 (dz*3+dy)][w' 0..33][ci 0..63], bf16
  __shared__ unsigned short slab[9 * 34 * 64];
  const int tid = threadIdx.x;
  const int bb  = blockIdx.z;
  const int t   = blockIdx.y / H_;
  const int h   = blockIdx.y % H_;
  const int w0  = blockIdx.x * 32;

  // --- stage slab via async global->LDS 16B block copies (halo -> zeros) ---
  {
    const unsigned lds_base = (unsigned)(unsigned long long)(void*)slab;
    for (int s = tid; s < 2448; s += 256) {        // 306 rows * 8 x 16B
      const int part = s & 7;
      const int row  = s >> 3;
      const int tap9 = row / 34;
      const int wp   = row % 34;
      const int dz = tap9 / 3, dy = tap9 % 3;
      const int tz = t + dz - 1, hy = h + dy - 1, wg = w0 - 1 + wp;
      const int eidx = ((tap9 * 34 + wp) << 6) + part * 8;   // ushort index in slab
      if ((unsigned)tz < T_ && (unsigned)hy < H_ && (unsigned)wg < W_) {
        const unsigned long long gaddr = (unsigned long long)(const void*)(
            xt + ((((size_t)bb * T_ + tz) * H_ + hy) * W_ + wg) * 64 + part * 8);
        const unsigned lds_off = lds_base + (unsigned)(eidx * 2);
        asm volatile("global_load_async_to_lds_b128 %0, %1, off"
                     :: "v"(lds_off), "v"(gaddr) : "memory");
      } else {
        v4u z = {0u, 0u, 0u, 0u};
        *(v4u*)&slab[eidx] = z;
      }
    }
    asm volatile("s_wait_asynccnt 0x0" ::: "memory");
  }
  __syncthreads();

  // --- WMMA main loop ---
  const int lane  = tid & 31;
  const int wv    = tid >> 5;
  const int oc0   = (wv & 3) * 16;          // 4 oc tiles
  const int pTile = wv >> 2;                // 2 position tiles
  const int m     = lane & 15;              // A row / B column
  const int hs    = lane >> 4;              // half-wave select
  const int lw    = pTile * 16 + m;         // local w (0..31)

  const unsigned short* arow =
      dynw + (size_t)(bb * 64 + oc0 + m) * KTOT + hs * 8;

  v8f c = {};
  for (int ks = 0; ks < 54; ++ks) {
    union { v16bf v; v4u u[2]; } a, bf;
    // A: two 16B chunks: K = ks*32 + hs*8 + {0..7} and +16..+23
    const v4u* ap = (const v4u*)(arow + ks * 32);
    a.u[0] = ap[0];
    a.u[1] = ap[2];
    // B: ci = ci0 + hs*8 + {0..7} and +16..+23 at (tap, w'+dx)
    const int tap27 = ks >> 1;
    const int ci0   = (ks & 1) << 5;
    const int dz = tap27 / 9, rem = tap27 % 9;
    const int dy = rem / 3, dx = rem % 3;
    const v4u* bp = (const v4u*)&slab[(((dz * 3 + dy) * 34 + (lw + dx)) << 6) + ci0 + hs * 8];
    bf.u[0] = bp[0];
    bf.u[1] = bp[2];
    c = __builtin_amdgcn_wmma_f32_16x16x32_bf16(false, a.v, false, bf.v,
                                                (short)0, c, false, false);
  }

  // --- epilogue: + dyn_b (vector-loaded), coalesced-offset stores of y ---
  const int w = w0 + lw;
  if (w < W_) {
    const int ocb = oc0 + hs * 8;                    // 8 consecutive oc per lane
    const float4* dbp = (const float4*)(dynb + bb * 64 + ocb);
    const float4 d0 = dbp[0];
    const float4 d1 = dbp[1];
    float bias[8] = {d0.x, d0.y, d0.z, d0.w, d1.x, d1.y, d1.z, d1.w};
    float* yp = y + ((((size_t)bb * 64 + ocb) * T_ + t) * H_ + h) * W_ + w;
#pragma unroll
    for (int r = 0; r < 8; ++r) {
      yp[(size_t)r * PC] = c[r] + bias[r];           // imm-offset stores, no movrels
    }
  }
}

// per-channel sum / sumsq over y
__global__ void k_stats(const float* __restrict__ y, float* __restrict__ stats) {
  __shared__ float s1[256], s2[256];
  const int oc = blockIdx.x, chunk = blockIdx.y;
  float a = 0.f, b = 0.f;
  for (int i = threadIdx.x; i < 1568; i += 256) {   // 8 samples * 196 float4
    const int bsel = i / 196, off = i % 196;
    const float4* p = (const float4*)(y + (size_t)(bsel * 64 + oc) * PC + chunk * 784);
    float4 v = p[off];
    a += v.x + v.y + v.z + v.w;
    b += v.x * v.x + v.y * v.y + v.z * v.z + v.w * v.w;
  }
  s1[threadIdx.x] = a; s2[threadIdx.x] = b;
  __syncthreads();
  for (int s = 128; s > 0; s >>= 1) {
    if ((int)threadIdx.x < s) {
      s1[threadIdx.x] += s1[threadIdx.x + s];
      s2[threadIdx.x] += s2[threadIdx.x + s];
    }
    __syncthreads();
  }
  if (threadIdx.x == 0) {
    atomicAdd(&stats[oc],      s1[0]);
    atomicAdd(&stats[64 + oc], s2[0]);
  }
}

// BN (batch stats, biased var) + ReLU -> out
__global__ void k_bn(const float* __restrict__ y, const float* __restrict__ stats,
                     const float* __restrict__ gamma, const float* __restrict__ beta,
                     float* __restrict__ out) {
  const int gid = blockIdx.x * 256 + threadIdx.x;   // exactly NELEM/4 threads
  const int oc  = (int)((((size_t)gid * 4) / PC) & 63);
  const float mean = stats[oc] * (1.0f / NRED);
  const float var  = stats[64 + oc] * (1.0f / NRED) - mean * mean;
  const float sc   = rsqrtf(var + 1e-5f) * gamma[oc];
  const float sh   = beta[oc] - mean * sc;
  float4 v = ((const float4*)y)[gid];
  float4 o;
  o.x = fmaxf(v.x * sc + sh, 0.f);
  o.y = fmaxf(v.y * sc + sh, 0.f);
  o.z = fmaxf(v.z * sc + sh, 0.f);
  o.w = fmaxf(v.w * sc + sh, 0.f);
  ((float4*)out)[gid] = o;
}

extern "C" void kernel_launch(void* const* d_in, const int* in_sizes, int n_in,
                              void* d_out, int out_size, void* d_ws, size_t ws_size,
                              hipStream_t stream) {
  const float* x       = (const float*)d_in[0];
  const float* weights = (const float*)d_in[1];
  const float* biases  = (const float*)d_in[2];
  const float* W1      = (const float*)d_in[3];
  const float* b1      = (const float*)d_in[4];
  const float* W2      = (const float*)d_in[5];
  const float* b2      = (const float*)d_in[6];
  const float* gamma   = (const float*)d_in[7];
  const float* beta    = (const float*)d_in[8];
  const int*   epochs  = (const int*)d_in[9];

  char* ws = (char*)d_ws;
  float*          y      = (float*)(ws + OFF_Y);
  unsigned short* dynw   = (unsigned short*)(ws + OFF_DYNW);
  float*          pooled = (float*)(ws + OFF_POOL);
  float*          phi    = (float*)(ws + OFF_PHI);
  float*          dynb   = (float*)(ws + OFF_DYNB);
  float*          stats  = (float*)(ws + OFF_STAT);
  unsigned short* xt     = (unsigned short*)(ws + OFF_XT);
  float*          out    = (float*)d_out;

  k_init<<<1, 128, 0, stream>>>(stats);
  k_maxpool<<<B_ * CIN, 256, 0, stream>>>(x, pooled);
  k_xt<<<B_ * T_ * H_, 256, 0, stream>>>(x, xt);
  k_mlp<<<1, 64, 0, stream>>>(pooled, W1, b1, W2, b2, biases, epochs, phi, dynb);
  k_dynw<<<3456, 256, 0, stream>>>(weights, phi, dynw);
  dim3 gc(2, T_ * H_, B_);
  k_conv<<<gc, 256, 0, stream>>>(xt, dynw, dynb, y);
  dim3 gs(64, 64);
  k_stats<<<gs, 256, 0, stream>>>(y, stats);
  k_bn<<<NELEM / 1024, 256, 0, stream>>>(y, stats, gamma, beta, out);
}
